// EGNN_layer_87643102642635
// MI455X (gfx1250) — matrix operand order repaired and checked by reading, not compiled, not run on hardware
//
#include <hip/hip_runtime.h>
#include <hip/hip_bf16.h>
#include <cstdint>

typedef __bf16 bf16_t;
typedef __attribute__((ext_vector_type(16))) __bf16 v16bf;
typedef __attribute__((ext_vector_type(8)))  float  v8f;

union Frag32B { uint4 q[2]; v16bf v; };

// A-matrix (16x32 bf16) fragment from row-major LDS tile.
// Per ISA 7.12.2: lane holds row M=lane%16; VGPR0..3 pack K = half*8 + {0..7},
// VGPR4..7 pack K = 16 + half*8 + {0..7}  (two contiguous 16B runs, 16 elems apart).
__device__ __forceinline__ v16bf load_a_frag_lds(const bf16_t* base, int ldk, int lane, int kstep) {
    const bf16_t* p = base + (size_t)(lane & 15) * ldk + kstep * 32 + (lane >> 4) * 8;
    Frag32B u;
    u.q[0] = ((const uint4*)p)[0];   // K run 1 (8 bf16)
    u.q[1] = ((const uint4*)p)[2];   // K run 2 (+16 elements)
    return u.v;
}

// B-matrix fragment, pre-packed lane-contiguous: 8 dwords per lane (32B).
__device__ __forceinline__ v16bf load_b_frag(const uint32_t* wp) {
    Frag32B u;
    u.q[0] = ((const uint4*)wp)[0];
    u.q[1] = ((const uint4*)wp)[1];
    return u.v;
}

// Fast activation math: v_exp_f32 + v_rcp_f32 (1-ulp HW approx), no IEEE div chains.
__device__ __forceinline__ float sigmoid_f(float z) {
    return __builtin_amdgcn_rcpf(1.0f + __expf(-z));
}
__device__ __forceinline__ float silu_f(float z) { return z * sigmoid_f(z); }

// Pack W[n, k] (row-major, row stride rowStride) into WMMA B-fragment layout:
// dst[((nb*ksteps + k)*32 + lane)*8 + v] packs bf16 pair at
// n = nb*16 + lane%16 ; k0 = k*32 + (lane/16)*8 + (v%4)*2 + (v/4)*16. 8 col-blocks fixed (128 out).
__global__ void pack_wmma_b(const float* __restrict__ W, int rowStride, int ksteps,
                            uint32_t* __restrict__ dst) {
    int t = blockIdx.x * blockDim.x + threadIdx.x;
    int per = ksteps * 256;
    int nb = t / per;
    int r  = t % per;
    int k  = r >> 8;
    int q  = r & 255;
    int lane = q >> 3;
    int v  = q & 7;
    int n  = nb * 16 + (lane & 15);
    int k0 = k * 32 + ((lane >> 4) << 3) + ((v & 3) << 1) + ((v >> 2) << 4);
    const float* wr = W + (size_t)n * rowStride + k0;
    union { bf16_t b; uint16_t u; } c0, c1;
    c0.b = (bf16_t)wr[0];
    c1.b = (bf16_t)wr[1];
    dst[t] = (uint32_t)c0.u | ((uint32_t)c1.u << 16);
}

// BatchNorm pass 1: per-column partial sum / sumsq, fp32 atomics into stats[0..255].
__global__ void bn_partial(const float* __restrict__ h, float* __restrict__ stats, int nnodes) {
    int c = threadIdx.x;                      // 128 threads: one column each
    int chunk = (nnodes + gridDim.x - 1) / gridDim.x;
    int r0 = blockIdx.x * chunk;
    int r1 = r0 + chunk; if (r1 > nnodes) r1 = nnodes;
    float s = 0.0f, s2 = 0.0f;
    for (int r = r0; r < r1; ++r) {
        float v = h[(size_t)r * 128 + c];
        s += v; s2 += v * v;
    }
    unsafeAtomicAdd(&stats[c], s);
    unsafeAtomicAdd(&stats[128 + c], s2);
}

// BatchNorm pass 2: normalize + affine, emit bf16 h_norm (L2-resident gather source).
__global__ void bn_apply(const float* __restrict__ h, const float* __restrict__ stats,
                         const float* __restrict__ gamma, const float* __restrict__ beta,
                         bf16_t* __restrict__ hnorm, int total, float inv_n) {
    int i = blockIdx.x * blockDim.x + threadIdx.x;
    if (i >= total) return;
    int c = i & 127;
    float mean = stats[c] * inv_n;
    float var  = stats[128 + c] * inv_n - mean * mean;
    float scale = gamma[c] * __builtin_amdgcn_rsqf(var + 1e-5f);
    hnorm[i] = (bf16_t)((h[i] - mean) * scale + beta[c]);
}

// Fused edge pipeline: gather -> f_e (2x GEMM+SiLU via WMMA) -> f_inf gate -> gated scatter-add.
// 8 waves/block; wave w owns output columns [16w, 16w+16); W fragments persistent in VGPRs.
__global__ __launch_bounds__(256) void egnn_edge_kernel(
    const bf16_t* __restrict__ hnorm, const float* __restrict__ x,
    const int* __restrict__ e, int nedges,
    const uint32_t* __restrict__ w1p, const uint32_t* __restrict__ w2p,
    const float* __restrict__ fe_w1, const float* __restrict__ fe_b1,
    const float* __restrict__ fe_b2,
    const float* __restrict__ finf_w, const float* __restrict__ finf_b,
    float* __restrict__ magg, int ntiles)
{
    __shared__ bf16_t featbuf[16 * 256];   // [h_src | h_dst] tile, 8KB
    __shared__ bf16_t m1buf[16 * 128];     // layer-1 activations, 4KB
    __shared__ float  distb[16];
    __shared__ int    srcb[16];
    __shared__ float  gateb[16];

    const int tid  = threadIdx.x;
    const int lane = tid & 31;
    const int wv   = tid >> 5;            // wave id == output column block
    const int hf   = lane >> 4;           // half-wave
    const int col  = wv * 16 + (lane & 15);
    const int row  = tid >> 4;            // staging row
    const int prt  = tid & 15;            // staging part

    // Persistent weight fragments (layer1: K=256 -> 8 steps; layer2: K=128 -> 4 steps)
    v16bf w1f[8], w2f[4];
#pragma unroll
    for (int k = 0; k < 8; ++k) w1f[k] = load_b_frag(w1p + (((size_t)wv * 8 + k) * 32 + lane) * 8);
#pragma unroll
    for (int k = 0; k < 4; ++k) w2f[k] = load_b_frag(w2p + (((size_t)wv * 4 + k) * 32 + lane) * 8);

    const float b1    = fe_b1[col];
    const float b2    = fe_b2[col];
    const float wdist = fe_w1[(size_t)col * 257 + 256];  // the dist column (K=256) of W1
    const float wg    = finf_w[col];
    const float gb    = finf_b[0];

    for (int tile = blockIdx.x; tile < ntiles; tile += gridDim.x) {
        __syncthreads();                   // guard LDS reuse across iterations
        const int ebase = tile * 16;
        if (tid < 16) {
            int s = e[ebase + tid];
            int d = e[nedges + ebase + tid];
            srcb[tid] = s;
            float dx = x[s * 3 + 0] - x[d * 3 + 0];
            float dy = x[s * 3 + 1] - x[d * 3 + 1];
            float dz = x[s * 3 + 2] - x[d * 3 + 2];
            distb[tid] = sqrtf(dx * dx + dy * dy + dz * dz);
            gateb[tid] = 0.0f;
        }
        {   // gather [h_src | h_dst] into LDS (bf16, 32B per thread)
            int s = e[ebase + row];
            int d = e[nedges + ebase + row];
            const bf16_t* sp = (prt < 8) ? (hnorm + (size_t)s * 128 + prt * 16)
                                         : (hnorm + (size_t)d * 128 + (prt - 8) * 16);
            uint4* o = (uint4*)(featbuf + row * 256 + prt * 16);
            o[0] = ((const uint4*)sp)[0];
            o[1] = ((const uint4*)sp)[1];
        }
        __syncthreads();

        // ---- layer 1: out = feat(16x256) @ W1^T + b1 + wdist*dist (rank-1) ----
        v8f acc;
#pragma unroll
        for (int v = 0; v < 8; ++v) acc[v] = b1 + wdist * distb[v + hf * 8];
#pragma unroll
        for (int k = 0; k < 8; ++k) {
            v16bf a = load_a_frag_lds(featbuf, 256, lane, k);
            acc = __builtin_amdgcn_wmma_f32_16x16x32_bf16(false, a, false, w1f[k],
                                                          (short)0, acc, false, false);
        }
#pragma unroll
        for (int v = 0; v < 8; ++v)
            m1buf[(v + hf * 8) * 128 + col] = (bf16_t)silu_f(acc[v]);
        __syncthreads();

        // ---- layer 2: m2 = silu(m1(16x128) @ W2^T + b2) ----
        v8f acc2;
#pragma unroll
        for (int v = 0; v < 8; ++v) acc2[v] = b2;
#pragma unroll
        for (int k = 0; k < 4; ++k) {
            v16bf a = load_a_frag_lds(m1buf, 128, lane, k);
            acc2 = __builtin_amdgcn_wmma_f32_16x16x32_bf16(false, a, false, w2f[k],
                                                           (short)0, acc2, false, false);
        }
        float m2[8];
#pragma unroll
        for (int v = 0; v < 8; ++v) m2[v] = silu_f(acc2[v]);

        // ---- gate: per-row dot(finf_w, m2); butterfly over 16-lane halves ----
#pragma unroll
        for (int v = 0; v < 8; ++v) {
            float t = m2[v] * wg;
            t += __shfl_xor(t, 1);
            t += __shfl_xor(t, 2);
            t += __shfl_xor(t, 4);
            t += __shfl_xor(t, 8);
            if ((lane & 15) == 0) atomicAdd(&gateb[v + hf * 8], t);
        }
        __syncthreads();

        // ---- gated scatter-sum into m_agg[src] ----
#pragma unroll
        for (int v = 0; v < 8; ++v) {
            int r = v + hf * 8;
            float g = sigmoid_f(gateb[r] + gb);
            unsafeAtomicAdd(&magg[(size_t)srcb[r] * 128 + col], g * m2[v]);
        }
    }
}

// Fused node update: u = silu([h_norm | m_i] @ W1^T + b1); out = h + u @ W2^T + b2.
__global__ __launch_bounds__(256) void egnn_node_kernel(
    const bf16_t* __restrict__ hnorm, const float* __restrict__ magg,
    const float* __restrict__ h0,
    const uint32_t* __restrict__ w1p, const uint32_t* __restrict__ w2p,
    const float* __restrict__ fh_b1, const float* __restrict__ fh_b2,
    float* __restrict__ out, int ntiles)
{
    __shared__ bf16_t featbuf[16 * 256];
    __shared__ bf16_t ubuf[16 * 128];

    const int tid  = threadIdx.x;
    const int lane = tid & 31;
    const int wv   = tid >> 5;
    const int hf   = lane >> 4;
    const int col  = wv * 16 + (lane & 15);
    const int row  = tid >> 4;
    const int prt  = tid & 15;

    v16bf w1f[8], w2f[4];
#pragma unroll
    for (int k = 0; k < 8; ++k) w1f[k] = load_b_frag(w1p + (((size_t)wv * 8 + k) * 32 + lane) * 8);
#pragma unroll
    for (int k = 0; k < 4; ++k) w2f[k] = load_b_frag(w2p + (((size_t)wv * 4 + k) * 32 + lane) * 8);
    const float b1 = fh_b1[col];
    const float b2 = fh_b2[col];

    for (int tile = blockIdx.x; tile < ntiles; tile += gridDim.x) {
        __syncthreads();
        {
            int node = tile * 16 + row;
            bf16_t* o = featbuf + row * 256 + prt * 16;
            if (prt < 8) {
                const uint4* i4 = (const uint4*)(hnorm + (size_t)node * 128 + prt * 16);
                ((uint4*)o)[0] = i4[0];
                ((uint4*)o)[1] = i4[1];
            } else {
                const float* mg = magg + (size_t)node * 128 + (prt - 8) * 16;
#pragma unroll
                for (int i = 0; i < 16; ++i) o[i] = (bf16_t)mg[i];
            }
        }
        __syncthreads();

        v8f acc;
#pragma unroll
        for (int v = 0; v < 8; ++v) acc[v] = b1;
#pragma unroll
        for (int k = 0; k < 8; ++k) {
            v16bf a = load_a_frag_lds(featbuf, 256, lane, k);
            acc = __builtin_amdgcn_wmma_f32_16x16x32_bf16(false, a, false, w1f[k],
                                                          (short)0, acc, false, false);
        }
#pragma unroll
        for (int v = 0; v < 8; ++v)
            ubuf[(v + hf * 8) * 128 + col] = (bf16_t)silu_f(acc[v]);
        __syncthreads();

        v8f acc2;
#pragma unroll
        for (int v = 0; v < 8; ++v) acc2[v] = b2;
#pragma unroll
        for (int k = 0; k < 4; ++k) {
            v16bf a = load_a_frag_lds(ubuf, 128, lane, k);
            acc2 = __builtin_amdgcn_wmma_f32_16x16x32_bf16(false, a, false, w2f[k],
                                                           (short)0, acc2, false, false);
        }
#pragma unroll
        for (int v = 0; v < 8; ++v) {
            int node = tile * 16 + v + hf * 8;
            size_t idx = (size_t)node * 128 + col;
            out[idx] = h0[idx] + acc2[v];   // residual
        }
    }
}

extern "C" void kernel_launch(void* const* d_in, const int* in_sizes, int n_in,
                              void* d_out, int out_size, void* d_ws, size_t ws_size,
                              hipStream_t stream) {
    const float* h      = (const float*)d_in[0];
    const float* x      = (const float*)d_in[1];
    const int*   e      = (const int*)d_in[2];
    const float* bn_g   = (const float*)d_in[3];
    const float* bn_b   = (const float*)d_in[4];
    const float* fe_w1  = (const float*)d_in[5];
    const float* fe_b1  = (const float*)d_in[6];
    const float* fe_w2  = (const float*)d_in[7];
    const float* fe_b2  = (const float*)d_in[8];
    const float* finf_w = (const float*)d_in[9];
    const float* finf_b = (const float*)d_in[10];
    const float* fh_w1  = (const float*)d_in[11];
    const float* fh_b1  = (const float*)d_in[12];
    const float* fh_w2  = (const float*)d_in[13];
    const float* fh_b2  = (const float*)d_in[14];

    const int nnodes = in_sizes[0] / 128;
    const int nedges = in_sizes[2] / 2;

    // Workspace carve-out (all 256B aligned)
    size_t off = 0;
    auto take = [&](size_t bytes) { size_t r = off; off += (bytes + 255) & ~(size_t)255; return r; };
    float*    stats = (float*)   ((char*)d_ws + take(256 * 4));
    bf16_t*   hnorm = (bf16_t*)  ((char*)d_ws + take((size_t)nnodes * 128 * 2));
    float*    magg  = (float*)   ((char*)d_ws + take((size_t)nnodes * 128 * 4));
    uint32_t* few1p = (uint32_t*)((char*)d_ws + take((size_t)8 * 8 * 256 * 4));
    uint32_t* few2p = (uint32_t*)((char*)d_ws + take((size_t)8 * 4 * 256 * 4));
    uint32_t* fhw1p = (uint32_t*)((char*)d_ws + take((size_t)8 * 8 * 256 * 4));
    uint32_t* fhw2p = (uint32_t*)((char*)d_ws + take((size_t)8 * 4 * 256 * 4));

    hipMemsetAsync(stats, 0, 256 * 4, stream);
    hipMemsetAsync(magg, 0, (size_t)nnodes * 128 * 4, stream);

    // Weight packing into WMMA B-fragment layout (bf16)
    pack_wmma_b<<<64, 256, 0, stream>>>(fe_w1, 257, 8, few1p);
    pack_wmma_b<<<32, 256, 0, stream>>>(fe_w2, 128, 4, few2p);
    pack_wmma_b<<<64, 256, 0, stream>>>(fh_w1, 256, 8, fhw1p);
    pack_wmma_b<<<32, 256, 0, stream>>>(fh_w2, 128, 4, fhw2p);

    // BatchNorm
    bn_partial<<<256, 128, 0, stream>>>(h, stats, nnodes);
    int total = nnodes * 128;
    bn_apply<<<(total + 255) / 256, 256, 0, stream>>>(h, stats, bn_g, bn_b, hnorm,
                                                      total, 1.0f / (float)nnodes);

    // Fused edge pipeline (16 edges per tile)
    int etiles = nedges / 16;
    int eblocks = etiles < 1024 ? etiles : 1024;
    egnn_edge_kernel<<<eblocks, 256, 0, stream>>>(hnorm, x, e, nedges, few1p, few2p,
                                                  fe_w1, fe_b1, fe_b2, finf_w, finf_b,
                                                  magg, etiles);

    // Fused node update (16 nodes per tile) -> d_out[0 : nnodes*128)
    int ntilesN = nnodes / 16;
    int nblocks = ntilesN < 512 ? ntilesN : 512;
    egnn_node_kernel<<<nblocks, 256, 0, stream>>>(hnorm, magg, h, fhw1p, fhw2p,
                                                  fh_b1, fh_b2, (float*)d_out, ntilesN);

    // Second tuple output: pass-through edge index array (int bits into the float tail)
    hipMemcpyAsync((char*)d_out + (size_t)nnodes * 128 * 4, e,
                   (size_t)nedges * 2 * 4, hipMemcpyDeviceToDevice, stream);

    (void)n_in; (void)out_size; (void)ws_size;
}
